// BetterGuidedAnchorHead_40819369181789
// MI455X (gfx1250) — matrix operand order repaired and checked by reading, not compiled, not run on hardware
//
#include <hip/hip_runtime.h>
#include <hip/hip_bf16.h>
#include <math.h>

typedef __attribute__((ext_vector_type(16))) _Float16 v16h;
typedef __attribute__((ext_vector_type(8)))  float    v8f;
typedef __attribute__((ext_vector_type(8)))  _Float16 h8;

#define IMGS 2
#define CIN  256
#define HH   112
#define WW_  112
#define HW   (HH * WW_)
#define NTAPS 9

// ---------------------------------------------------------------------------
// CDNA5 async global->LDS copy (ASYNCcnt). vdst VGPR = per-lane LDS byte addr
// (low 32 bits of generic shared pointer = LDS offset per aperture rules),
// vaddr = 64-bit global address. Tracked with s_wait_asynccnt.
// ---------------------------------------------------------------------------
__device__ __forceinline__ void async_b128(void* lds, const void* g) {
  unsigned l = (unsigned)(size_t)lds;
  asm volatile("global_load_async_to_lds_b128 %0, %1, off"
               :: "v"(l), "v"(g) : "memory");
}
__device__ __forceinline__ void wait_async0() {
  asm volatile("s_wait_asynccnt 0x0" ::: "memory");
}
__device__ __forceinline__ void wait_async4() {  // keep newest 4 in flight
  asm volatile("s_wait_asynccnt 0x4" ::: "memory");
}

// ---------------------------------------------------------------------------
// WMMA 16x16x32 f16 fragment index helpers (wave32, ISA 7.12.2)
// ---------------------------------------------------------------------------
__device__ __forceinline__ int kA_idx(int v, int lane) {
  return ((v & 4) ? 16 : 0) + ((lane & 16) ? 8 : 0) + 2 * (v & 3);
}
__device__ __forceinline__ unsigned pack2h(float lo, float hi) {
  union { unsigned u; _Float16 h[2]; } p;
  p.h[0] = (_Float16)lo; p.h[1] = (_Float16)hi;
  return p.u;
}
union FragU { v16h v; unsigned u[8]; };

// ---------------------------------------------------------------------------
// NCHW fp32 -> NHWC f16 (LDS transpose, coalesced both sides)
// ---------------------------------------------------------------------------
__global__ __launch_bounds__(256) void nchw_to_nhwc_h_kernel(
    const float* __restrict__ in, _Float16* __restrict__ outh) {
  __shared__ __align__(16) _Float16 L[16][CIN];
  const int t = threadIdx.x;
  const int mt = blockIdx.x;
  const int img = mt / (HW / 16);
  const int pix0 = (mt % (HW / 16)) * 16;
  {
    const int p = t & 15;
    const float* src = in + (size_t)img * CIN * HW + pix0 + p;
#pragma unroll
    for (int i = 0; i < 16; ++i) {
      const int c = i * 16 + (t >> 4);
      L[p][c] = (_Float16)src[(size_t)c * HW];
    }
  }
  __syncthreads();
  {
    const int p = t & 15, cb = (t >> 4) * 16;
    _Float16* dst = outh + ((size_t)img * HW + pix0 + p) * CIN + cb;
    *(h8*)dst = *(const h8*)&L[p][cb];
    *(h8*)(dst + 8) = *(const h8*)&L[p][cb + 8];
  }
}

// ---------------------------------------------------------------------------
// Weight packing: w [O=256][C=256][3][3] fp32 -> wp [tap 9][cpair 128][O 256]
// ---------------------------------------------------------------------------
__global__ void pack3x3_kernel(const float* __restrict__ w, unsigned* __restrict__ wp) {
  int idx = blockIdx.x * blockDim.x + threadIdx.x;
  if (idx >= 9 * 128 * 256) return;
  int o = idx & 255;
  int rest = idx >> 8;
  int cp = rest & 127;
  int t = rest >> 7;
  float lo = w[((size_t)o * 256 + 2 * cp) * 9 + t];
  float hi = w[((size_t)o * 256 + 2 * cp + 1) * 9 + t];
  wp[((size_t)t * 128 + cp) * 256 + o] = pack2h(lo, hi);
}

__global__ void pack_head19_kernel(const float* __restrict__ wl, const float* __restrict__ bl,
                                   const float* __restrict__ wpt, const float* __restrict__ bpt,
                                   unsigned* __restrict__ wp, float* __restrict__ bias) {
  int idx = blockIdx.x * blockDim.x + threadIdx.x;
  if (idx >= 128 * 32) return;
  int o = idx & 31;
  int cp = idx >> 5;
  float lo = 0.f, hi = 0.f;
  if (o == 0)      { lo = wl[2 * cp];                  hi = wl[2 * cp + 1]; }
  else if (o < 19) { lo = wpt[(o - 1) * 256 + 2 * cp]; hi = wpt[(o - 1) * 256 + 2 * cp + 1]; }
  wp[(size_t)cp * 32 + o] = pack2h(lo, hi);
  if (idx < 32) bias[idx] = (idx == 0) ? bl[0] : (idx < 19 ? bpt[idx - 1] : 0.f);
}

__global__ void pack_head98_kernel(const float* __restrict__ wc, const float* __restrict__ bc,
                                   const float* __restrict__ wr, const float* __restrict__ br,
                                   unsigned* __restrict__ wp, float* __restrict__ bias) {
  int idx = blockIdx.x * blockDim.x + threadIdx.x;
  if (idx >= 128 * 128) return;
  int o = idx & 127;
  int cp = idx >> 7;
  float lo = 0.f, hi = 0.f;
  if (o < 80)      { lo = wc[(size_t)o * 256 + 2 * cp];        hi = wc[(size_t)o * 256 + 2 * cp + 1]; }
  else if (o < 98) { lo = wr[(size_t)(o - 80) * 256 + 2 * cp]; hi = wr[(size_t)(o - 80) * 256 + 2 * cp + 1]; }
  wp[(size_t)cp * 128 + o] = pack2h(lo, hi);
  if (idx < 128) bias[idx] = (idx < 80) ? bc[idx] : (idx < 98 ? br[idx - 80] : 0.f);
}

// ---------------------------------------------------------------------------
// conv3x3 + bias + ReLU. NHWC f16 in/out. Async double-buffered weight tiles.
// Block: 8 waves; M = 16 px of one row, N = 128 couts; K = 9 taps x 256 cin.
// ---------------------------------------------------------------------------
__global__ __launch_bounds__(256) void conv3x3_wmma_kernel(
    const _Float16* __restrict__ xh, const unsigned* __restrict__ wp,
    const float* __restrict__ bias, _Float16* __restrict__ outh) {
  __shared__ __align__(16) _Float16 As[16][CIN];      // 8 KB
  __shared__ __align__(16) unsigned Bsp[2][32][128];  // 2 x 16 KB

  const int tid = threadIdx.x, lane = tid & 31, wv = tid >> 5;
  int mt = blockIdx.x;
  const int img = mt / (HH * 7); mt -= img * (HH * 7);
  const int y = mt / 7;
  const int x0 = (mt % 7) * 16;
  const int o0 = blockIdx.y * 128;

  auto issueB = [&](int q) {  // chunk q = tap*4+cc : 32 cin-pairs x 128 couts
    const unsigned* wsrc = wp + ((size_t)(q >> 2) * 128 + 32 * (q & 3)) * 256 + o0;
    unsigned(*dst)[128] = Bsp[q & 1];
#pragma unroll
    for (int j = 0; j < 4; ++j) {
      const int idx = j * 256 + tid;          // b128 transfer id, 1024 total
      const int kp = idx >> 5, o4 = (idx & 31) * 4;
      async_b128(&dst[kp][o4], wsrc + (size_t)kp * 256 + o4);
    }
  };

  v8f acc = {};
  issueB(0);
  for (int q = 0; q < 36; ++q) {
    const int t = q >> 2, cc = q & 3;
    __syncthreads();  // all waves done with buf[(q+1)&1] and As from prior iters
    if (cc == 0) {    // stage A for tap t: 16 px x 256 cin, async when in-bounds
      const int ky = t / 3 - 1, kx = t % 3 - 1;
      const int yy = y + ky;
      const int p = tid & 15, cb = (tid >> 4) * 16;
      const int xx = x0 + p + kx;
      if (yy >= 0 && yy < HH && xx >= 0 && xx < WW_) {
        const _Float16* src = xh + ((size_t)img * HW + (size_t)yy * WW_ + xx) * CIN + cb;
        async_b128(&As[p][cb], src);
        async_b128(&As[p][cb + 8], src + 8);
      } else {
        h8 z = (_Float16)0.f;
        *(h8*)&As[p][cb] = z;
        *(h8*)&As[p][cb + 8] = z;
      }
    }
    if (q + 1 < 36) { issueB(q + 1); wait_async4(); }  // chunk q + A complete
    else            { wait_async0(); }
    __syncthreads();
    const unsigned(*bb)[128] = Bsp[q & 1];
#pragma unroll
    for (int kk = 0; kk < 2; ++kk) {
      FragU a, b;
      const int cbase = cc * 64 + kk * 32;
#pragma unroll
      for (int v = 0; v < 8; ++v) {
        a.u[v] = *(const unsigned*)&As[lane & 15][cbase + kA_idx(v, lane)];
        b.u[v] = bb[kk * 16 + (kA_idx(v, lane) >> 1)][wv * 16 + (lane & 15)];
      }
      acc = __builtin_amdgcn_wmma_f32_16x16x32_f16(false, a.v, false, b.v,
                                                   (short)0, acc, false, false);
    }
  }
  const int n = lane & 15, o = o0 + wv * 16 + n;
  const float bv = bias[o];
#pragma unroll
  for (int v = 0; v < 8; ++v) {
    const int pm = v + ((lane & 16) ? 8 : 0);
    float r = acc[v] + bv;
    r = r > 0.f ? r : 0.f;
    outh[((size_t)img * HW + (size_t)y * WW_ + x0 + pm) * CIN + o] = (_Float16)r;
  }
}

// ---------------------------------------------------------------------------
// DCNv1 + ReLU. Sample pos = (y + pts_y, x + pts_x); pts in headA[pix][1..18].
// Bilinear blend as packed f16 FMA over contiguous NHWC rows.
// ---------------------------------------------------------------------------
__global__ __launch_bounds__(256) void dcn_wmma_kernel(
    const _Float16* __restrict__ feat_h, const float* __restrict__ headA,
    const unsigned* __restrict__ wp, _Float16* __restrict__ outh) {
  __shared__ __align__(16) _Float16 As[16][CIN];
  __shared__ __align__(16) unsigned Bsp[2][32][128];
  __shared__ int   i00[16], i01[16], i10[16], i11[16];
  __shared__ float q00[16], q01[16], q10[16], q11[16];

  const int tid = threadIdx.x, lane = tid & 31, wv = tid >> 5;
  int mt = blockIdx.x;
  const int img = mt / (HH * 7); mt -= img * (HH * 7);
  const int y = mt / 7;
  const int x0 = (mt % 7) * 16;
  const int o0 = blockIdx.y * 128;

  auto issueB = [&](int q) {
    const unsigned* wsrc = wp + ((size_t)(q >> 2) * 128 + 32 * (q & 3)) * 256 + o0;
    unsigned(*dst)[128] = Bsp[q & 1];
#pragma unroll
    for (int j = 0; j < 4; ++j) {
      const int idx = j * 256 + tid;
      const int kp = idx >> 5, o4 = (idx & 31) * 4;
      async_b128(&dst[kp][o4], wsrc + (size_t)kp * 256 + o4);
    }
  };

  v8f acc = {};
  issueB(0);
  for (int q = 0; q < 36; ++q) {
    const int t = q >> 2, cc = q & 3;
    __syncthreads();
    if (cc == 0) {
      if (tid < 16) {  // per-position bilinear params for tap t
        const int p = tid, xx = x0 + p;
        const float* ha = headA + ((size_t)img * HW + (size_t)y * WW_ + xx) * 19;
        const float py = (float)y + ha[1 + 2 * t];
        const float px = (float)xx + ha[2 + 2 * t];
        const float y0f = floorf(py), x0f = floorf(px);
        const int yi = (int)y0f, xi = (int)x0f;
        const float dy = py - y0f, dx = px - x0f;
        const int y0c = min(max(yi, 0), HH - 1),  y1c = min(max(yi + 1, 0), HH - 1);
        const int x0c = min(max(xi, 0), WW_ - 1), x1c = min(max(xi + 1, 0), WW_ - 1);
        const float vy0 = (yi >= 0 && yi < HH) ? 1.f : 0.f;
        const float vy1 = (yi + 1 >= 0 && yi + 1 < HH) ? 1.f : 0.f;
        const float vx0 = (xi >= 0 && xi < WW_) ? 1.f : 0.f;
        const float vx1 = (xi + 1 >= 0 && xi + 1 < WW_) ? 1.f : 0.f;
        i00[p] = y0c * WW_ + x0c;  q00[p] = (1.f - dy) * (1.f - dx) * vy0 * vx0;
        i01[p] = y0c * WW_ + x1c;  q01[p] = (1.f - dy) * dx * vy0 * vx1;
        i10[p] = y1c * WW_ + x0c;  q10[p] = dy * (1.f - dx) * vy1 * vx0;
        i11[p] = y1c * WW_ + x1c;  q11[p] = dy * dx * vy1 * vx1;
      }
      __syncthreads();
      {  // sample 16 px x 256 cin via 4-corner packed-f16 FMA
        const int p = tid & 15, cb = (tid >> 4) * 16;
        const _Float16* fb = feat_h + (size_t)img * HW * CIN;
        const _Float16* r0 = fb + (size_t)i00[p] * CIN + cb;
        const _Float16* r1 = fb + (size_t)i01[p] * CIN + cb;
        const _Float16* r2 = fb + (size_t)i10[p] * CIN + cb;
        const _Float16* r3 = fb + (size_t)i11[p] * CIN + cb;
        const h8 w0 = (_Float16)q00[p], w1 = (_Float16)q01[p];
        const h8 w2 = (_Float16)q10[p], w3 = (_Float16)q11[p];
        h8 lo = w0 * *(const h8*)r0 + w1 * *(const h8*)r1
              + w2 * *(const h8*)r2 + w3 * *(const h8*)r3;
        h8 hi = w0 * *(const h8*)(r0 + 8) + w1 * *(const h8*)(r1 + 8)
              + w2 * *(const h8*)(r2 + 8) + w3 * *(const h8*)(r3 + 8);
        *(h8*)&As[p][cb] = lo;
        *(h8*)&As[p][cb + 8] = hi;
      }
    }
    if (q + 1 < 36) { issueB(q + 1); wait_async4(); }
    else            { wait_async0(); }
    __syncthreads();
    const unsigned(*bb)[128] = Bsp[q & 1];
#pragma unroll
    for (int kk = 0; kk < 2; ++kk) {
      FragU a, b;
      const int cbase = cc * 64 + kk * 32;
#pragma unroll
      for (int v = 0; v < 8; ++v) {
        a.u[v] = *(const unsigned*)&As[lane & 15][cbase + kA_idx(v, lane)];
        b.u[v] = bb[kk * 16 + (kA_idx(v, lane) >> 1)][wv * 16 + (lane & 15)];
      }
      acc = __builtin_amdgcn_wmma_f32_16x16x32_f16(false, a.v, false, b.v,
                                                   (short)0, acc, false, false);
    }
  }
  const int n = lane & 15, o = o0 + wv * 16 + n;
#pragma unroll
  for (int v = 0; v < 8; ++v) {
    const int pm = v + ((lane & 16) ? 8 : 0);
    float r = acc[v];
    r = r > 0.f ? r : 0.f;
    outh[((size_t)img * HW + (size_t)y * WW_ + x0 + pm) * CIN + o] = (_Float16)r;
  }
}

// ---------------------------------------------------------------------------
// Generic 1x1 conv (NHWC f16 in -> pixel-major fp32 [pix][ncout] out)
// ---------------------------------------------------------------------------
__global__ __launch_bounds__(256) void conv1x1_wmma_kernel(
    const _Float16* __restrict__ inh, const unsigned* __restrict__ wp,
    const float* __restrict__ bias, float* __restrict__ out,
    int ncout, int npad) {
  __shared__ __align__(16) _Float16 As[16][CIN];   // 8 KB
  __shared__ __align__(16) unsigned Bsp[16][128];  // 8 KB

  const int tid = threadIdx.x, lane = tid & 31, wv = tid >> 5;
  const int mt = blockIdx.x;
  const int img = mt / (HW / 16);
  const int pix0 = (mt % (HW / 16)) * 16;

  {  // A stage: 16 px x 256 cin, contiguous async b128
    const int p = tid & 15, cb = (tid >> 4) * 16;
    const _Float16* src = inh + ((size_t)img * HW + pix0 + p) * CIN + cb;
    async_b128(&As[p][cb], src);
    async_b128(&As[p][cb + 8], src + 8);
  }

  v8f acc = {};
  const bool active = (wv * 16) < npad;
  for (int cc = 0; cc < 8; ++cc) {
    __syncthreads();
    for (int idx = tid; idx < 4 * npad; idx += 256) {  // b128 transfers
      const int u = idx * 4;
      const int kp = u / npad, o4 = u - kp * npad;
      async_b128(&Bsp[kp][o4], wp + (size_t)(cc * 16 + kp) * npad + o4);
    }
    wait_async0();
    __syncthreads();
    if (active) {
      FragU a, b;
#pragma unroll
      for (int v = 0; v < 8; ++v) {
        a.u[v] = *(const unsigned*)&As[lane & 15][cc * 32 + kA_idx(v, lane)];
        b.u[v] = Bsp[kA_idx(v, lane) >> 1][wv * 16 + (lane & 15)];
      }
      acc = __builtin_amdgcn_wmma_f32_16x16x32_f16(false, a.v, false, b.v,
                                                   (short)0, acc, false, false);
    }
  }
  if (active) {
    const int n = lane & 15, o = wv * 16 + n;
    if (o < ncout) {
      const float bv = bias[o];
#pragma unroll
      for (int v = 0; v < 8; ++v) {
        const int pm = v + ((lane & 16) ? 8 : 0);
        out[((size_t)img * HW + pix0 + pm) * ncout + o] = acc[v] + bv;
      }
    }
  }
}

// ---------------------------------------------------------------------------
// Assemble [N,117,H,W]: loc | pts_init | masked cls | masked refine + pts_init
// Heads are pixel-major: headA [img][pix][19], headB [img][pix][98].
// ---------------------------------------------------------------------------
__global__ void assemble_kernel(const float* __restrict__ headA,
                                const float* __restrict__ headB,
                                float* __restrict__ out) {
  const int idx = blockIdx.x * blockDim.x + threadIdx.x;
  if (idx >= IMGS * HW) return;
  const int img = idx / HW, pix = idx - img * HW;
  const float* hA = headA + ((size_t)img * HW + pix) * 19;
  const float* hB = headB + ((size_t)img * HW + pix) * 98;
  float* o = out + (size_t)img * 117 * HW;

  o[pix] = hA[0];                                       // loc_pred
#pragma unroll
  for (int t = 0; t < 18; ++t)                           // pts_init
    o[(size_t)(1 + t) * HW + pix] = hA[1 + t];

  const bool mask = headA[(size_t)pix * 19] >= -4.595119850134589f;  // img0 loc
#pragma unroll 4
  for (int c = 0; c < 80; ++c)                           // masked cls
    o[(size_t)(19 + c) * HW + pix] = mask ? hB[c] : 0.f;
#pragma unroll
  for (int t = 0; t < 18; ++t) {                         // masked refine + pts
    const float pr = mask ? hB[80 + t] : 0.f;
    o[(size_t)(99 + t) * HW + pix] = pr + hA[1 + t];
  }
}

// ---------------------------------------------------------------------------
extern "C" void kernel_launch(void* const* d_in, const int* in_sizes, int n_in,
                              void* d_out, int out_size, void* d_ws, size_t ws_size,
                              hipStream_t stream) {
  const float* x     = (const float*)d_in[0];
  const float* w_ref = (const float*)d_in[1];
  const float* b_ref = (const float*)d_in[2];
  const float* w_loc = (const float*)d_in[3];
  const float* b_loc = (const float*)d_in[4];
  const float* w_pts = (const float*)d_in[5];
  const float* b_pts = (const float*)d_in[6];
  const float* w_dcn = (const float*)d_in[7];
  const float* w_cls = (const float*)d_in[8];
  const float* b_cls = (const float*)d_in[9];
  const float* w_pr  = (const float*)d_in[10];
  const float* b_pr  = (const float*)d_in[11];
  float* out = (float*)d_out;

  char* ws = (char*)d_ws;
  size_t off = 0;
  auto alloc = [&](size_t bytes) -> void* {
    void* p = ws + off;
    off += (bytes + 255) & ~(size_t)255;
    return p;
  };
  _Float16* xh     = (_Float16*)alloc((size_t)IMGS * HW * CIN * 2);  // 12.8 MB
  _Float16* feat_h = (_Float16*)alloc((size_t)IMGS * HW * CIN * 2);  // 12.8 MB
  _Float16* xa_h   = (_Float16*)alloc((size_t)IMGS * HW * CIN * 2);  // 12.8 MB
  float*    headA  = (float*)alloc((size_t)IMGS * HW * 19 * 4);
  float*    headB  = (float*)alloc((size_t)IMGS * HW * 98 * 4);
  unsigned* wp1    = (unsigned*)alloc((size_t)9 * 128 * 256 * 4);
  unsigned* wp2    = (unsigned*)alloc((size_t)9 * 128 * 256 * 4);
  unsigned* wp3    = (unsigned*)alloc((size_t)128 * 32 * 4);
  unsigned* wp4    = (unsigned*)alloc((size_t)128 * 128 * 4);
  float*    bias3  = (float*)alloc(32 * 4);
  float*    bias4  = (float*)alloc(128 * 4);
  (void)ws_size; (void)in_sizes; (void)n_in; (void)out_size;

  nchw_to_nhwc_h_kernel<<<IMGS * (HW / 16), 256, 0, stream>>>(x, xh);
  pack3x3_kernel<<<(9 * 128 * 256 + 255) / 256, 256, 0, stream>>>(w_ref, wp1);
  pack3x3_kernel<<<(9 * 128 * 256 + 255) / 256, 256, 0, stream>>>(w_dcn, wp2);
  pack_head19_kernel<<<(128 * 32 + 255) / 256, 256, 0, stream>>>(w_loc, b_loc, w_pts, b_pts, wp3, bias3);
  pack_head98_kernel<<<(128 * 128 + 255) / 256, 256, 0, stream>>>(w_cls, b_cls, w_pr, b_pr, wp4, bias4);

  const dim3 gconv(IMGS * HH * (WW_ / 16), 2);
  conv3x3_wmma_kernel<<<gconv, 256, 0, stream>>>(xh, wp1, b_ref, feat_h);
  conv1x1_wmma_kernel<<<IMGS * (HW / 16), 256, 0, stream>>>(feat_h, wp3, bias3, headA, 19, 32);
  dcn_wmma_kernel<<<gconv, 256, 0, stream>>>(feat_h, headA, wp2, xa_h);
  conv1x1_wmma_kernel<<<IMGS * (HW / 16), 256, 0, stream>>>(xa_h, wp4, bias4, headB, 98, 128);
  assemble_kernel<<<(IMGS * HW + 255) / 256, 256, 0, stream>>>(headA, headB, out);
}
